// TransformerLayer_46669114638423
// MI455X (gfx1250) — matrix operand order repaired and checked by reading, not compiled, not run on hardware
//
#include <hip/hip_runtime.h>
#include <math.h>

// ---------------------------------------------------------------------------
// Problem constants (match reference)
// ---------------------------------------------------------------------------
#define H_    2048
#define NQ_   16
#define NKV_  4
#define HD_   128
#define REP_  4
#define FF_   8192
#define B_    2
#define S_    2048
#define M_    (B_ * S_)       // 4096 tokens
#define KV_   (NKV_ * HD_)    // 512
#define EPS_  1e-6f

typedef __attribute__((ext_vector_type(16))) __bf16         v16bf;
typedef __attribute__((ext_vector_type(16))) unsigned short v16us;
typedef __attribute__((ext_vector_type(8)))  float          v8f;

// ---------------------------------------------------------------------------
// Helpers
// ---------------------------------------------------------------------------
static __device__ __forceinline__ unsigned short f2bf(float f) {
  union { float f; unsigned int u; } v; v.f = f;
  unsigned int u = v.u;
  u += 0x7FFFu + ((u >> 16) & 1u);           // round-to-nearest-even
  return (unsigned short)(u >> 16);
}

static __device__ __forceinline__ v8f zero8() {
  v8f z;
#pragma unroll
  for (int e = 0; e < 8; ++e) z[e] = 0.0f;
  return z;
}

// Async global -> LDS copy of 16 bytes per lane (CDNA5 async path, ASYNCcnt).
// LDS offset = low 32 bits of the generic pointer.
static __device__ __forceinline__ void g2l_b128(unsigned short* lds_dst,
                                                const unsigned short* gsrc) {
  asm volatile("global_load_async_to_lds_b128 %0, %1, off"
               :: "v"((unsigned)(unsigned long long)lds_dst), "v"(gsrc)
               : "memory");
}
static __device__ __forceinline__ void wait_async0() {
  asm volatile("s_wait_asynccnt 0x0" ::: "memory");
}
// Wait until at most 8 async ops outstanding (per-wave, in-order completion:
// releases after the first batch of 8 is done while the next 8 stay in flight).
static __device__ __forceinline__ void wait_async8() {
  asm volatile("s_wait_asynccnt 0x8" ::: "memory");
}

// A fragment: 16x32 bf16, row-major LDS (lda elems). Two contiguous 16B chunks.
// Lanes 0-15: row M=lane, K {kh..kh+7} U {kh+16..kh+23}, kh=0; lanes 16-31: kh=8.
static __device__ __forceinline__ v16bf frag_av(const unsigned short* base, int lda, int lane) {
  const int row = lane & 15;
  const int kh  = (lane >> 4) << 3;
  const unsigned short* p = base + row * lda + kh;
  v16us u;
  *(uint4*)&u         = *(const uint4*)(p);
  *(((uint4*)&u) + 1) = *(const uint4*)(p + 16);
  return __builtin_bit_cast(v16bf, u);
}

// B fragment (32x16) from K-contiguous LDS layout T[n][k] (ld = k stride).
// Lane col = lane%16, K {kb..kb+7} U {kb+16..kb+23}, kb=(lane/16)*8 -> two 16B chunks.
static __device__ __forceinline__ v16bf frag_btv(const unsigned short* base, int ld, int lane) {
  const int col = lane & 15;
  const int kb  = (lane >> 4) << 3;
  const unsigned short* p = base + col * ld + kb;
  v16us u;
  *(uint4*)&u         = *(const uint4*)(p);
  *(((uint4*)&u) + 1) = *(const uint4*)(p + 16);
  return __builtin_bit_cast(v16bf, u);
}

static __device__ __forceinline__ v8f wmma_bf16(v16bf a, v16bf b, v8f c) {
  return __builtin_amdgcn_wmma_f32_16x16x32_bf16(false, a, false, b, (short)0, c, false, false);
}

// ---------------------------------------------------------------------------
// Weight conversion: fp32 [K][N] -> bf16 TRANSPOSED [N][K], tiled through LDS
// so both global streams stay coalesced. Grid: (N/32, K/32), 256 threads.
// ---------------------------------------------------------------------------
__global__ __launch_bounds__(256) void cvt_tr_kernel(const float* __restrict__ in,
                                                     unsigned short* __restrict__ out,
                                                     int K, int N) {
  __shared__ float tile[32][33];
  const int tx = threadIdx.x & 31;
  const int ty = threadIdx.x >> 5;                 // 0..7
  const int k0 = blockIdx.y * 32;
  const int n0 = blockIdx.x * 32;
#pragma unroll
  for (int j = 0; j < 32; j += 8)
    tile[ty + j][tx] = in[(size_t)(k0 + ty + j) * N + n0 + tx];
  __syncthreads();
#pragma unroll
  for (int j = 0; j < 32; j += 8)
    out[(size_t)(n0 + ty + j) * K + k0 + tx] = f2bf(tile[tx][ty + j]);
}

// ---------------------------------------------------------------------------
// RMSNorm: one block per token, fp32 in -> bf16 out
// ---------------------------------------------------------------------------
__global__ __launch_bounds__(256) void rmsnorm_kernel(const float* __restrict__ x,
                                                      const float* __restrict__ g,
                                                      unsigned short* __restrict__ out) {
  __shared__ float red[256];
  const int row = blockIdx.x;
  const float* xr = x + (size_t)row * H_;
  float ss = 0.0f;
  for (int i = threadIdx.x; i < H_; i += 256) { float v = xr[i]; ss += v * v; }
  red[threadIdx.x] = ss;
  __syncthreads();
  for (int s = 128; s > 0; s >>= 1) {
    if (threadIdx.x < s) red[threadIdx.x] += red[threadIdx.x + s];
    __syncthreads();
  }
  const float rinv = rsqrtf(red[0] * (1.0f / H_) + EPS_);
  for (int i = threadIdx.x; i < H_; i += 256)
    out[(size_t)row * H_ + i] = f2bf(xr[i] * rinv * g[i]);
}

// ---------------------------------------------------------------------------
// GEMM: C[M,N] = A[M,K](bf16) * Bt[N,K](bf16, pre-transposed) + bias.
// Epilogue modes:
//   mode 0: store bf16
//   mode 1: store f32 (acc + bias + resid)
//   mode 2: store bf16 of exact GELU(acc + bias)
//   mode 3: store bf16 transposed as Vt[b][kvh][d][s]  (for attention V)
// Block: 256 threads, 128x128 tile, BK=64. 8 waves -> 32x64 each.
// Double-buffered LDS: async DMA of slab t+1 overlaps WMMA on slab t.
// ---------------------------------------------------------------------------
__global__ __launch_bounds__(256) void gemm_kernel(const unsigned short* __restrict__ A,
                                                   const unsigned short* __restrict__ Bt,
                                                   const float* __restrict__ bias,
                                                   const float* __restrict__ resid,
                                                   void* __restrict__ Cout,
                                                   int M, int N, int K, int mode) {
  __shared__ __align__(16) unsigned short As[2][128 * 64];    // [m][k]
  __shared__ __align__(16) unsigned short BsT[2][128 * 64];   // [n][k]
  const int tid   = threadIdx.x;
  const int lane  = tid & 31;
  const int wave  = tid >> 5;
  const int m0    = (wave >> 1) * 32;
  const int n0    = (wave & 1) * 64;
  const int mBase = blockIdx.y * 128;
  const int nBase = blockIdx.x * 128;

  auto issue_tile = [&](int buf, int kbb) {
#pragma unroll
    for (int j = 0; j < 4; ++j) {
      const int e = (tid + j * 256) * 8;
      const int r = e >> 6, c = e & 63;      // both tiles are 128 x 64
      g2l_b128(&As[buf][e],  A  + (size_t)(mBase + r) * K + kbb + c);
      g2l_b128(&BsT[buf][e], Bt + (size_t)(nBase + r) * K + kbb + c);
    }
  };

  v8f acc[2][4];
#pragma unroll
  for (int i = 0; i < 2; ++i)
#pragma unroll
    for (int j = 0; j < 4; ++j) acc[i][j] = zero8();

  issue_tile(0, 0);
  wait_async0();
  __syncthreads();

  int cur = 0;
  for (int kb = 0; kb < K; kb += 64) {
    const int nxt = cur ^ 1;
    if (kb + 64 < K) {
      issue_tile(nxt, kb + 64);              // DMA next slab, overlapped with WMMA
      __builtin_prefetch(A  + (size_t)(mBase + (tid >> 1)) * K + kb + 128 + (tid & 1) * 32, 0, 0);
      __builtin_prefetch(Bt + (size_t)(nBase + (tid >> 1)) * K + kb + 128 + (tid & 1) * 32, 0, 0);
    }
#pragma unroll
    for (int ks = 0; ks < 64; ks += 32) {
      v16bf af[2], bfr[4];
#pragma unroll
      for (int mt = 0; mt < 2; ++mt)
        af[mt] = frag_av(&As[cur][(m0 + mt * 16) * 64 + ks], 64, lane);
#pragma unroll
      for (int nt = 0; nt < 4; ++nt)
        bfr[nt] = frag_btv(&BsT[cur][(n0 + nt * 16) * 64 + ks], 64, lane);
#pragma unroll
      for (int mt = 0; mt < 2; ++mt)
#pragma unroll
        for (int nt = 0; nt < 4; ++nt)
          acc[mt][nt] = wmma_bf16(af[mt], bfr[nt], acc[mt][nt]);
    }
    wait_async0();
    __syncthreads();
    cur = nxt;
  }

  // Epilogue. C layout: VGPR r -> M = r (lanes 0-15) / r+8 (lanes 16-31), N = lane%16.
  const int halfOff = (lane >> 4) << 3;
  const int colOff  = lane & 15;
#pragma unroll
  for (int mt = 0; mt < 2; ++mt)
#pragma unroll
    for (int nt = 0; nt < 4; ++nt)
#pragma unroll
      for (int r = 0; r < 8; ++r) {
        const int gm = mBase + m0 + mt * 16 + r + halfOff;
        const int gn = nBase + n0 + nt * 16 + colOff;
        if (gm < M && gn < N) {
          float v = acc[mt][nt][r] + bias[gn];
          const size_t idx = (size_t)gm * N + gn;
          if (mode == 0) {
            ((unsigned short*)Cout)[idx] = f2bf(v);
          } else if (mode == 1) {
            ((float*)Cout)[idx] = v + resid[idx];
          } else if (mode == 2) {
            const float gv = 0.5f * v * (1.0f + erff(v * 0.70710678118654752f));
            ((unsigned short*)Cout)[idx] = f2bf(gv);
          } else {
            // mode 3: transposed V store: [b][kvh][d][s]
            const int bb  = gm >> 11;            // gm / S_
            const int ssp = gm & (S_ - 1);
            const int kvh = gn >> 7;             // gn / HD_
            const int dd  = gn & (HD_ - 1);
            ((unsigned short*)Cout)[(((size_t)bb * NKV_ + kvh) * HD_ + dd) * S_ + ssp] = f2bf(v);
          }
        }
      }
}

// ---------------------------------------------------------------------------
// GQA causal flash attention. One block = (batch b, head h, 64-query tile).
// 4 waves; wave w owns query rows w*16..w*16+15. Keys streamed 64 at a time.
// Separate K ([key][hd]) and V ([hd][key], pre-transposed in global) buffers:
//   - V(kb) DMA overlaps score dump + softmax (wait_asynccnt<=8 releases it),
//   - K(kb+1) DMA overlaps softmax + PV (drained by wait_asynccnt 0 at tail).
// ---------------------------------------------------------------------------
__global__ __launch_bounds__(128) void attn_kernel(const unsigned short* __restrict__ Q,
                                                   const unsigned short* __restrict__ Kc,
                                                   const unsigned short* __restrict__ Vt,
                                                   unsigned short* __restrict__ Ctx) {
  __shared__ __align__(16) unsigned short Qs[64 * 128];   // 16 KB  [q][hd]
  __shared__ __align__(16) unsigned short Ks[64 * 128];   // 16 KB  [key][hd]
  __shared__ __align__(16) unsigned short Vs[128 * 64];   // 16 KB  [hd][key]
  __shared__ float          Ss[4][16 * 64];               // 16 KB  raw scores (f32)
  __shared__ __align__(4) unsigned short Ps[4][16 * 64];  //  8 KB  probabilities (bf16)
  __shared__ float m_s[64], l_s[64], fac_s[64];

  const int tid  = threadIdx.x;
  const int lane = tid & 31;
  const int wave = tid >> 5;
  const int nqb  = S_ / 64;
  const int qb   = blockIdx.x % nqb;
  const int h    = (blockIdx.x / nqb) % NQ_;
  const int b    = blockIdx.x / (nqb * NQ_);
  const int kvh  = h / REP_;
  const int qBase = qb * 64;
  const float scale = 0.08838834764831845f;   // 1/sqrt(128)

  const int halfOff = (lane >> 4) << 3;
  const int colOff  = lane & 15;

  auto issue_k = [&](int kB) {
#pragma unroll
    for (int j = 0; j < 8; ++j) {
      const int e = (tid + j * 128) * 8;
      const int r = e >> 7, col = e & 127;
      g2l_b128(Ks + e, Kc + (size_t)(b * S_ + kB + r) * KV_ + kvh * HD_ + col);
    }
  };
  auto issue_v = [&](int kB) {
#pragma unroll
    for (int j = 0; j < 8; ++j) {
      const int e = (tid + j * 128) * 8;
      const int d = e >> 6, kc = e & 63;     // e == d*64 + kc
      g2l_b128(Vs + e, Vt + ((size_t)(b * NKV_ + kvh) * HD_ + d) * S_ + kB + kc);
    }
  };

  // prologue: Q block + K(0)
#pragma unroll
  for (int j = 0; j < 8; ++j) {
    const int e = (tid + j * 128) * 8;
    const int r = e >> 7, col = e & 127;
    g2l_b128(Qs + e, Q + (size_t)(b * S_ + qBase + r) * H_ + h * HD_ + col);
  }
  issue_k(0);
  if (tid < 64) { m_s[tid] = -1e30f; l_s[tid] = 0.0f; }
  v8f ctx[8];
#pragma unroll
  for (int i = 0; i < 8; ++i) ctx[i] = zero8();
  wait_async0();
  __syncthreads();

  for (int kb = 0; kb <= qb; ++kb) {
    const int kBase = kb * 64;

    // --- scores: 16 rows x 64 keys, K-dim = HD = 128 (from Ks) ---
    v8f sc[4];
#pragma unroll
    for (int i = 0; i < 4; ++i) sc[i] = zero8();
#pragma unroll
    for (int ks = 0; ks < 128; ks += 32) {
      const v16bf aq = frag_av(Qs + (wave * 16) * 128 + ks, 128, lane);
#pragma unroll
      for (int nt = 0; nt < 4; ++nt) {
        const v16bf bk = frag_btv(Ks + (nt * 16) * 128 + ks, 128, lane);
        sc[nt] = wmma_bf16(aq, bk, sc[nt]);
      }
    }
    issue_v(kBase);                          // V DMA overlaps dump + softmax
#pragma unroll
    for (int nt = 0; nt < 4; ++nt)
#pragma unroll
      for (int r = 0; r < 8; ++r)
        Ss[wave][(r + halfOff) * 64 + nt * 16 + colOff] = sc[nt][r];
    __syncthreads();

    // all waves finished reading Ks -> prefetch next K block (overlaps softmax+PV)
    const int kNext = (kb < qb) ? kBase + 64 : kBase;
    issue_k(kNext);

    // --- online softmax: 2 lanes per row, 32 keys each ---
    {
      const int rl   = lane >> 1;
      const int half = lane & 1;
      const int row  = wave * 16 + rl;
      const int qg   = qBase + row;
      const float* sp = &Ss[wave][rl * 64 + half * 32];
      float vals[32];
      float mloc = -1e30f;
#pragma unroll
      for (int j = 0; j < 32; ++j) {
        const int kg = kBase + half * 32 + j;
        float s = sp[j] * scale;
        if (kg > qg) s = -1e30f;              // causal mask
        vals[j] = s;
        mloc = fmaxf(mloc, s);
      }
      mloc = fmaxf(mloc, __shfl_xor(mloc, 1, 32));
      const float mold = m_s[row];
      const float mnew = fmaxf(mold, mloc);
      const float fac  = __expf(mold - mnew);
      float sum = 0.0f;
      unsigned* pp32 = (unsigned*)&Ps[wave][rl * 64 + half * 32];
#pragma unroll
      for (int j = 0; j < 16; ++j) {
        const float p0 = __expf(vals[2 * j]     - mnew);
        const float p1 = __expf(vals[2 * j + 1] - mnew);
        sum += p0 + p1;
        pp32[j] = (unsigned)f2bf(p0) | ((unsigned)f2bf(p1) << 16);
      }
      sum += __shfl_xor(sum, 1, 32);
      if (half == 0) {
        m_s[row]   = mnew;
        l_s[row]   = l_s[row] * fac + sum;
        fac_s[row] = fac;
      }
    }
    __syncthreads();

    // --- rescale running ctx ---
#pragma unroll
    for (int r = 0; r < 8; ++r) {
      const float f = fac_s[wave * 16 + r + halfOff];
#pragma unroll
      for (int nt = 0; nt < 8; ++nt) ctx[nt][r] *= f;
    }
    wait_async8();                           // V landed (next-K may still be in flight)
    __syncthreads();

    // --- ctx += P(16x64) @ V(64x128), V fragments k-contiguous ---
#pragma unroll
    for (int ks = 0; ks < 64; ks += 32) {
      const v16bf ap = frag_av(&Ps[wave][ks], 64, lane);
#pragma unroll
      for (int nt = 0; nt < 8; ++nt) {
        const v16bf bv = frag_btv(Vs + (nt * 16) * 64 + ks, 64, lane);
        ctx[nt] = wmma_bf16(ap, bv, ctx[nt]);
      }
    }
    wait_async0();                           // next-K landed
    __syncthreads();
  }

  // --- write normalized context, bf16, layout [b, s, h*HD + d] ---
#pragma unroll
  for (int nt = 0; nt < 8; ++nt)
#pragma unroll
    for (int r = 0; r < 8; ++r) {
      const int row  = wave * 16 + r + halfOff;
      const float li = 1.0f / l_s[row];
      const int q = qBase + row;
      const int n = nt * 16 + colOff;
      Ctx[(size_t)(b * S_ + q) * H_ + h * HD_ + n] = f2bf(ctx[nt][r] * li);
    }
}

// ---------------------------------------------------------------------------
// Host launch
// ---------------------------------------------------------------------------
extern "C" void kernel_launch(void* const* d_in, const int* in_sizes, int n_in,
                              void* d_out, int out_size, void* d_ws, size_t ws_size,
                              hipStream_t stream) {
  (void)in_sizes; (void)n_in; (void)out_size; (void)ws_size;
  const float* x  = (const float*)d_in[0];
  const float* Wq = (const float*)d_in[1];
  const float* bq = (const float*)d_in[2];
  const float* Wk = (const float*)d_in[3];
  const float* bk = (const float*)d_in[4];
  const float* Wv = (const float*)d_in[5];
  const float* bv = (const float*)d_in[6];
  const float* Wo = (const float*)d_in[7];
  const float* bo = (const float*)d_in[8];
  const float* W1 = (const float*)d_in[9];
  const float* b1 = (const float*)d_in[10];
  const float* W2 = (const float*)d_in[11];
  const float* b2 = (const float*)d_in[12];
  const float* g1 = (const float*)d_in[13];
  const float* g2 = (const float*)d_in[14];
  float* out = (float*)d_out;

  char* ws = (char*)d_ws;
  size_t o = 0;
  auto alloc = [&](size_t bytes) -> void* {
    void* p = ws + o;
    o += (bytes + 255) & ~(size_t)255;
    return p;
  };
  // bf16 transposed weights [N][K]
  unsigned short* WqT = (unsigned short*)alloc((size_t)H_ * H_ * 2);
  unsigned short* WkT = (unsigned short*)alloc((size_t)H_ * KV_ * 2);
  unsigned short* WvT = (unsigned short*)alloc((size_t)H_ * KV_ * 2);
  unsigned short* WoT = (unsigned short*)alloc((size_t)H_ * H_ * 2);
  unsigned short* W1T = (unsigned short*)alloc((size_t)H_ * FF_ * 2);
  unsigned short* W2T = (unsigned short*)alloc((size_t)FF_ * H_ * 2);
  unsigned short* xn1 = (unsigned short*)alloc((size_t)M_ * H_ * 2);
  unsigned short* Qb  = (unsigned short*)alloc((size_t)M_ * H_ * 2);
  unsigned short* Kb  = (unsigned short*)alloc((size_t)M_ * KV_ * 2);
  unsigned short* Vtb = (unsigned short*)alloc((size_t)M_ * KV_ * 2);  // [b][kvh][d][s]
  unsigned short* ctxb= (unsigned short*)alloc((size_t)M_ * H_ * 2);
  float*          out1= (float*)alloc((size_t)M_ * H_ * 4);
  unsigned short* xn2 = (unsigned short*)alloc((size_t)M_ * H_ * 2);
  unsigned short* act = (unsigned short*)alloc((size_t)M_ * FF_ * 2);

  // weight conversion fp32 -> bf16, transposed [N][K]
  cvt_tr_kernel<<<dim3(H_ / 32,  H_ / 32),  256, 0, stream>>>(Wq, WqT, H_,  H_);
  cvt_tr_kernel<<<dim3(KV_ / 32, H_ / 32),  256, 0, stream>>>(Wk, WkT, H_,  KV_);
  cvt_tr_kernel<<<dim3(KV_ / 32, H_ / 32),  256, 0, stream>>>(Wv, WvT, H_,  KV_);
  cvt_tr_kernel<<<dim3(H_ / 32,  H_ / 32),  256, 0, stream>>>(Wo, WoT, H_,  H_);
  cvt_tr_kernel<<<dim3(FF_ / 32, H_ / 32),  256, 0, stream>>>(W1, W1T, H_,  FF_);
  cvt_tr_kernel<<<dim3(H_ / 32,  FF_ / 32), 256, 0, stream>>>(W2, W2T, FF_, H_);

  // pre-norm attention block
  rmsnorm_kernel<<<M_, 256, 0, stream>>>(x, g1, xn1);
  gemm_kernel<<<dim3(H_ / 128, M_ / 128), 256, 0, stream>>>(xn1, WqT, bq, nullptr, Qb, M_, H_, H_, 0);
  gemm_kernel<<<dim3(KV_ / 128, M_ / 128), 256, 0, stream>>>(xn1, WkT, bk, nullptr, Kb, M_, KV_, H_, 0);
  gemm_kernel<<<dim3(KV_ / 128, M_ / 128), 256, 0, stream>>>(xn1, WvT, bv, nullptr, Vtb, M_, KV_, H_, 3);
  attn_kernel<<<B_ * NQ_ * (S_ / 64), 128, 0, stream>>>(Qb, Kb, Vtb, ctxb);
  gemm_kernel<<<dim3(H_ / 128, M_ / 128), 256, 0, stream>>>(ctxb, WoT, bo, x, out1, M_, H_, H_, 1);

  // pre-norm FFN block
  rmsnorm_kernel<<<M_, 256, 0, stream>>>(out1, g2, xn2);
  gemm_kernel<<<dim3(FF_ / 128, M_ / 128), 256, 0, stream>>>(xn2, W1T, b1, nullptr, act, M_, FF_, H_, 2);
  gemm_kernel<<<dim3(H_ / 128, M_ / 128), 256, 0, stream>>>(act, W2T, b2, out1, out, M_, H_, FF_, 1);
}
